// SortingAttention_30202210025960
// MI455X (gfx1250) — matrix operand order repaired and checked by reading, not compiled, not run on hardware
//
#include <hip/hip_runtime.h>

// SortingAttention == counting-sort emitter for 16-bit keys.
// Phase 1: WMMA-based key extraction + global-atomic histogram (read 256 MB).
// Phase 2: 65536-bin exclusive scan (single workgroup, trivial).
// Phase 3: emit count[v] copies of v's bit pattern (write 256 MB).
// HBM floor ~512 MB / 23.3 TB/s ~= 22 us; everything else is noise.

typedef __attribute__((ext_vector_type(16))) _Float16 v16h;
typedef __attribute__((ext_vector_type(8)))  float    v8f;
typedef __attribute__((ext_vector_type(4)))  float    v4f;

#define NBINS 65536

// ---------------------------------------------------------------- zero counts
__global__ void sa_zero_kernel(unsigned* __restrict__ counts) {
    int i = blockIdx.x * blockDim.x + threadIdx.x;
    if (i < NBINS) counts[i] = 0u;
}

// ---------------------------------------------------------------- histogram
// One wave processes a 16-row tile per iteration:
//   keys(16 rows) = A(16x32 f16, bits, K=16..31 zero) x B(32x16 f16, pow2 weights)
// via v_wmma_f32_16x16x32_f16. Exact: bits in {0,1}, weights 2^0..2^15, f32 acc.
__global__ void sa_hist_kernel(const float* __restrict__ tokens,
                               unsigned* __restrict__ counts,
                               int num_tiles) {
    const int lane   = threadIdx.x & 31;
    const int wave   = (blockIdx.x * blockDim.x + threadIdx.x) >> 5;
    const int nwaves = (gridDim.x * blockDim.x) >> 5;

    // B matrix (32x16): elems 0..7 hold K=8*half+e -> weight 2^(15-K) (MSB first);
    // elems 8..15 hold K=16..31 -> zero (K-padding).
    v16h b;
#pragma unroll
    for (int k = 0; k < 16; ++k)
        b[k] = (k < 8) ? (_Float16)(float)(1u << (15 - (8 * (lane >> 4) + k))) : (_Float16)0.0f;

    // A layout (16x32, 16-bit): lane L -> row (L&15);
    // lanes 0-15 carry K=0..7 in elems 0..7, lanes 16-31 carry K=8..15.
    // Elems 8..15 map to K=16..31 -> keep zero.
    const int m    = lane & 15;
    const int half = lane >> 4;   // 0: bits 0..7, 1: bits 8..15

    for (int t = wave; t < num_tiles; t += nwaves) {
        const float* src = tokens + ((size_t)t * 16 + m) * 16 + half * 8;
        v4f f0 = *(const v4f*)(src);
        v4f f1 = *(const v4f*)(src + 4);

        // Prefetch next tile this wave will touch (global_prefetch_b8).
        if (t + nwaves < num_tiles)
            __builtin_prefetch(src + (size_t)nwaves * 256, 0, 1);

        v16h a = {};  // K=16..31 stay zero
#pragma unroll
        for (int j = 0; j < 4; ++j) {
            a[j]     = (_Float16)f0[j];
            a[4 + j] = (_Float16)f1[j];
        }

        v8f c = {};
        // (neg_a, A, neg_b, B, c_mod, C, reuse_a, reuse_b)
        c = __builtin_amdgcn_wmma_f32_16x16x32_f16(
                false, a, false, b, (short)0, c, false, false);

        // C layout: VGPR i, lanes 0-15 -> M=i (N=lane), lanes 16-31 -> M=i+8.
        // So lane 0 holds keys of rows 0..7 at N=0, lane 16 holds rows 8..15.
        // Only those two lanes update -> exactly one increment per row.
        if ((lane & 15) == 0) {
#pragma unroll
            for (int i = 0; i < 8; ++i) {
                unsigned key = (unsigned)(c[i] + 0.5f);  // exact integer in [0,65535]
                atomicAdd(&counts[key], 1u);             // L2 atomic, 64K bins spread
            }
        }
    }
}

// ---------------------------------------------------------------- scan (64K)
__global__ void sa_scan_kernel(const unsigned* __restrict__ counts,
                               unsigned* __restrict__ offsets) {
    __shared__ unsigned partial[256];
    __shared__ unsigned pref[256];
    const int t = threadIdx.x;

    unsigned s = 0;
    for (int j = 0; j < 256; ++j) s += counts[t * 256 + j];
    partial[t] = s;
    __syncthreads();

    if (t == 0) {
        unsigned run = 0;
        for (int i = 0; i < 256; ++i) { pref[i] = run; run += partial[i]; }
    }
    __syncthreads();

    unsigned run = pref[t];
    for (int j = 0; j < 256; ++j) {
        offsets[t * 256 + j] = run;
        run += counts[t * 256 + j];
    }
    if (t == 255) offsets[NBINS] = run;  // == SEQ_LEN
}

// ---------------------------------------------------------------- fill output
// For each key v, write count[v] identical 64-byte rows (bit pattern of v).
// One wave per key iteration: 32 lanes x float4 = 8 rows per store round.
__global__ void sa_fill_kernel(const unsigned* __restrict__ offsets,
                               float* __restrict__ out) {
    const int lane   = threadIdx.x & 31;
    const int wave   = (blockIdx.x * blockDim.x + threadIdx.x) >> 5;
    const int nwaves = (gridDim.x * blockDim.x) >> 5;

    const int p0  = (4 * lane) & 15;  // this lane's float-offset within a row
    const int sub = lane >> 2;        // row offset within an 8-row chunk

    for (int v = wave; v < NBINS; v += nwaves) {
        const unsigned beg = offsets[v];
        const unsigned cnt = offsets[v + 1] - beg;
        if (!cnt) continue;

        // Row element j of key v (MSB first) = bit (15 - j).
        v4f pat;
#pragma unroll
        for (int k = 0; k < 4; ++k)
            pat[k] = (float)((v >> (15 - (p0 + k))) & 1u);

        float* dst = out + (size_t)beg * 16 + lane * 4;  // 16B-aligned b128 stores
        for (unsigned r = 0; r < cnt; r += 8) {
            if (r + sub < cnt) *(v4f*)dst = pat;
            dst += 128;  // advance 8 rows
        }
    }
}

// ---------------------------------------------------------------- launch
extern "C" void kernel_launch(void* const* d_in, const int* in_sizes, int n_in,
                              void* d_out, int out_size, void* d_ws, size_t ws_size,
                              hipStream_t stream) {
    const float* tokens = (const float*)d_in[0];
    float*       out    = (float*)d_out;

    // Workspace: counts[65536] then offsets[65537] (u32); ~512 KB total.
    unsigned* counts  = (unsigned*)d_ws;
    unsigned* offsets = counts + NBINS;

    const int seq       = in_sizes[0] / 16;  // 4,194,304 rows
    const int num_tiles = seq / 16;          // 16 rows per WMMA tile

    sa_zero_kernel<<<(NBINS + 255) / 256, 256, 0, stream>>>(counts);
    sa_hist_kernel<<<2048, 256, 0, stream>>>(tokens, counts, num_tiles);
    sa_scan_kernel<<<1, 256, 0, stream>>>(counts, offsets);
    sa_fill_kernel<<<2048, 256, 0, stream>>>(offsets, out);
}
